// Qwen3MoeSparseMoeBlock_18038862643526
// MI455X (gfx1250) — matrix operand order repaired and checked by reading, not compiled, not run on hardware
//
#include <hip/hip_runtime.h>
#include <hip/hip_bf16.h>

// ---------------------------------------------------------------------------
// Qwen3 MoE sparse block for MI455X (gfx1250, wave32).
//   T = 8192 tokens, H = 2048, E = 8 experts, top-2 routing.
// v3: expert GEMM with register-level blocking: 128x64 block tile, 256
// threads = 8 waves in a 4x2 grid, each wave owns a 32x32 macro-tile
// (4 accumulators).  Per K-step: 8 ds_load_b128 -> 4 v_wmma_f32_16x16x32_bf16
// (2 loads/WMMA, half of v2's LDS pressure).  A and B staged in LDS as bf16;
// expert weights stay L2-resident (128 MB in 192 MB L2); total L2->WGP
// traffic ~6 GB.
// ---------------------------------------------------------------------------

#define MOE_T 8192
#define MOE_H 2048
#define MOE_E 8

#define MT 128         // gathered-token rows per block
#define NT 64          // output columns per block
#define KC 128         // K-chunk staged in LDS
#define THREADS 256    // 8 waves: 4 (M) x 2 (N) grid of 32x32 macro-tiles

typedef __attribute__((ext_vector_type(16))) __bf16 v16bf;
typedef __attribute__((ext_vector_type(8)))  __bf16 v8bf;
typedef __attribute__((ext_vector_type(4)))  __bf16 v4bf;
typedef __attribute__((ext_vector_type(8)))  float  v8f;

// fp32x4 global -> bf16x4 LDS (ds_store_b64), used for A and B staging
__device__ __forceinline__ void stage4(__bf16* dst, const float* src) {
    const float4 v = *(const float4*)src;
    v4bf p;
    p[0] = (__bf16)v.x; p[1] = (__bf16)v.y;
    p[2] = (__bf16)v.z; p[3] = (__bf16)v.w;
    *(v4bf*)dst = p;
}

// A fragment, CDNA5 ISA 7.12.2: lane row = lane&15, half h = lane>>4;
// elements = K(8h..8h+7) ++ K(16+8h..23+8h)   (two ds_load_b128)
__device__ __forceinline__ v16bf fragA(const __bf16* rowp, int k0, int h) {
    const v8bf x0 = *(const v8bf*)(rowp + k0 + 8 * h);
    const v8bf x1 = *(const v8bf*)(rowp + k0 + 16 + 8 * h);
    return __builtin_shufflevector(
        x0, x1, 0, 1, 2, 3, 4, 5, 6, 7, 8, 9, 10, 11, 12, 13, 14, 15);
}

// B fragment: lane col = lane&15; elements = K(8h..8h+7) ++ K(16+8h..23+8h) (two ds_load_b128)
__device__ __forceinline__ v16bf fragB(const __bf16* colp, int k0, int h) {
    const v8bf x0 = *(const v8bf*)(colp + k0 + 8 * h);
    const v8bf x1 = *(const v8bf*)(colp + k0 + 16 + 8 * h);
    return __builtin_shufflevector(
        x0, x1, 0, 1, 2, 3, 4, 5, 6, 7, 8, 9, 10, 11, 12, 13, 14, 15);
}

// ---------------------------------------------------------------------------
// Kernel 1: router.  One wave (32 lanes) per token.
// ---------------------------------------------------------------------------
__global__ __launch_bounds__(256) void moe_router_kernel(
    const float* __restrict__ x,        // [T, H]
    const float* __restrict__ gate_w,   // [E, H]
    float* __restrict__ logits_out,     // [T, E]
    int*   __restrict__ cnt,            // [E]
    int*   __restrict__ tok_idx,        // [E, T]
    float* __restrict__ tok_w)          // [E, T]
{
    const int wave = threadIdx.x >> 5;
    const int lane = threadIdx.x & 31;
    const int t = blockIdx.x * 8 + wave;
    if (t >= MOE_T) return;

    float acc[MOE_E];
#pragma unroll
    for (int e = 0; e < MOE_E; ++e) acc[e] = 0.0f;

    const float* xr = x + (size_t)t * MOE_H;
    for (int i = lane; i < MOE_H; i += 32) {
        const float xv = xr[i];
#pragma unroll
        for (int e = 0; e < MOE_E; ++e)
            acc[e] += xv * gate_w[e * MOE_H + i];
    }
#pragma unroll
    for (int e = 0; e < MOE_E; ++e) {
#pragma unroll
        for (int off = 16; off > 0; off >>= 1)
            acc[e] += __shfl_xor(acc[e], off, 32);
    }

    if (lane == 0) {
#pragma unroll
        for (int e = 0; e < MOE_E; ++e)
            logits_out[(size_t)t * MOE_E + e] = acc[e];

        int i0 = 0;
#pragma unroll
        for (int e = 1; e < MOE_E; ++e) if (acc[e] > acc[i0]) i0 = e;
        int i1 = (i0 == 0) ? 1 : 0;
#pragma unroll
        for (int e = 0; e < MOE_E; ++e)
            if (e != i0 && acc[e] > acc[i1]) i1 = e;

        const float m   = acc[i0];
        const float e1  = __expf(acc[i1] - m);
        const float inv = 1.0f / (1.0f + e1);
        const float w0  = inv;
        const float w1  = e1 * inv;

        int p0 = atomicAdd(&cnt[i0], 1);
        tok_idx[i0 * MOE_T + p0] = t;
        tok_w  [i0 * MOE_T + p0] = w0;
        int p1 = atomicAdd(&cnt[i1], 1);
        tok_idx[i1 * MOE_T + p1] = t;
        tok_w  [i1 * MOE_T + p1] = w1;
    }
}

// ---------------------------------------------------------------------------
// Kernel 2: expert GEMM, 128x64 block tile, 32x32 register macro-tiles.
// ---------------------------------------------------------------------------
__global__ __launch_bounds__(THREADS) void moe_expert_kernel(
    const float* __restrict__ x,          // [T, H]
    const float* __restrict__ expert_w,   // [E, H, H] (row = out feature)
    const int*   __restrict__ cnt,        // [E]
    const int*   __restrict__ tok_idx,    // [E, T]
    const float* __restrict__ tok_w,      // [E, T]
    float* __restrict__ out)              // [T, H], pre-zeroed
{
    const int e  = blockIdx.z;
    const int mt = blockIdx.y;              // MT-token tile within expert list
    const int nb = blockIdx.x * NT;         // output-column base

    const int count = cnt[e];
    if (mt * MT >= count) return;           // block-uniform early exit

    __shared__ __bf16 sA[MT][KC];           // 32 KB gathered-token rows
    __shared__ __bf16 sB[NT][KC];           // 16 KB W_e rows (out columns)
    __shared__ int    sTok[MT];
    __shared__ float  sW[MT];

    if (threadIdx.x < MT) {
        const int r = mt * MT + threadIdx.x;
        if (r < count) {
            sTok[threadIdx.x] = tok_idx[e * MOE_T + r];
            sW  [threadIdx.x] = tok_w  [e * MOE_T + r];
        } else {
            sTok[threadIdx.x] = 0;          // pad rows: weight 0 => no effect
            sW  [threadIdx.x] = 0.0f;
        }
    }

    const int wave  = threadIdx.x >> 5;
    const int lane  = threadIdx.x & 31;
    const int mwave = wave >> 1;            // 0..3 : 32-row macro-tile
    const int nwave = wave & 1;             // 0..1 : 32-col macro-tile
    const int l15   = lane & 15;
    const int h     = lane >> 4;            // K-half selector

    const int rowA0 = mwave * 32 + l15;     // A rows for the two M sub-tiles
    const int rowA1 = rowA0 + 16;
    const int colB0 = nwave * 32 + l15;     // B rows (out cols), two N sub-tiles
    const int colB1 = colB0 + 16;

    const float* wbase = expert_w + (size_t)e * MOE_H * MOE_H;

    v8f c00 = {}, c01 = {}, c10 = {}, c11 = {};

    for (int kk = 0; kk < MOE_H; kk += KC) {
        __syncthreads();                    // chunk (and sTok/sW) ready/retired

        // ---- stage A: MT x KC fp32 -> bf16 --------------------------------
        for (int q = threadIdx.x; q < (MT * KC) / 4; q += THREADS) {
            const int elem = q * 4;
            const int r = elem >> 7;                // / KC
            const int k = elem & (KC - 1);
            const float* src = x + (size_t)sTok[r] * MOE_H + kk + k;
            __builtin_prefetch(src + KC, 0, 0);     // next K-chunk
            stage4(&sA[r][k], src);
        }
        // ---- stage B: NT x KC fp32 -> bf16 --------------------------------
        for (int q = threadIdx.x; q < (NT * KC) / 4; q += THREADS) {
            const int elem = q * 4;
            const int r = elem >> 7;
            const int k = elem & (KC - 1);
            const float* src = wbase + (size_t)(nb + r) * MOE_H + kk + k;
            __builtin_prefetch(src + KC, 0, 0);
            stage4(&sB[r][k], src);
        }
        __syncthreads();

        // ---- K loop: 8 ds_load_b128 -> 4 WMMA per 32 K --------------------
#pragma unroll
        for (int k0 = 0; k0 < KC; k0 += 32) {
            const v16bf a0 = fragA(&sA[rowA0][0], k0, h);
            const v16bf a1 = fragA(&sA[rowA1][0], k0, h);
            const v16bf b0 = fragB(&sB[colB0][0], k0, h);
            const v16bf b1 = fragB(&sB[colB1][0], k0, h);

            c00 = __builtin_amdgcn_wmma_f32_16x16x32_bf16(
                      false, a0, false, b0, (short)0, c00, false, false);
            c01 = __builtin_amdgcn_wmma_f32_16x16x32_bf16(
                      false, a0, false, b1, (short)0, c01, false, false);
            c10 = __builtin_amdgcn_wmma_f32_16x16x32_bf16(
                      false, a1, false, b0, (short)0, c10, false, false);
            c11 = __builtin_amdgcn_wmma_f32_16x16x32_bf16(
                      false, a1, false, b1, (short)0, c11, false, false);
        }
    }

    // ---- weighted scatter-accumulate into out -----------------------------
    // C/D layout: n = lane&15 (tile col), m = vi + 8h (tile row)
    const int n0 = nb + colB0;
    const int n1 = nb + colB1;
    const int mBase0 = mwave * 32;          // rows of M sub-tile 0
    const int mBase1 = mwave * 32 + 16;     // rows of M sub-tile 1
#pragma unroll
    for (int vi = 0; vi < 8; ++vi) {
        const int m0 = mBase0 + vi + 8 * h;
        const int m1 = mBase1 + vi + 8 * h;
        const float w0 = sW[m0];
        const float w1 = sW[m1];
        float* d0 = out + (size_t)sTok[m0] * MOE_H;
        float* d1 = out + (size_t)sTok[m1] * MOE_H;
        __hip_atomic_fetch_add(d0 + n0, w0 * c00[vi], __ATOMIC_RELAXED,
                               __HIP_MEMORY_SCOPE_AGENT);
        __hip_atomic_fetch_add(d0 + n1, w0 * c01[vi], __ATOMIC_RELAXED,
                               __HIP_MEMORY_SCOPE_AGENT);
        __hip_atomic_fetch_add(d1 + n0, w1 * c10[vi], __ATOMIC_RELAXED,
                               __HIP_MEMORY_SCOPE_AGENT);
        __hip_atomic_fetch_add(d1 + n1, w1 * c11[vi], __ATOMIC_RELAXED,
                               __HIP_MEMORY_SCOPE_AGENT);
    }
}

// ---------------------------------------------------------------------------
// Launch wrapper. d_out = [T*H out | T*E logits] fp32.
// d_ws layout: [0..31] cnt (E ints), [256..] tok_idx (E*T ints),
//              then tok_w (E*T floats).  Total < 520 KB.
// ---------------------------------------------------------------------------
extern "C" void kernel_launch(void* const* d_in, const int* in_sizes, int n_in,
                              void* d_out, int out_size, void* d_ws, size_t ws_size,
                              hipStream_t stream) {
    (void)in_sizes; (void)n_in; (void)ws_size;

    const float* x        = (const float*)d_in[0];   // [4,2048,2048] fp32
    const float* gate_w   = (const float*)d_in[1];   // [8,2048] fp32
    const float* expert_w = (const float*)d_in[2];   // [8,2048,2048] fp32

    float* out    = (float*)d_out;                   // [T*H]
    float* logits = out + (size_t)MOE_T * MOE_H;     // [T*E]

    char*  ws      = (char*)d_ws;
    int*   cnt     = (int*)ws;
    int*   tok_idx = (int*)(ws + 256);
    float* tok_w   = (float*)(ws + 256 + sizeof(int) * MOE_E * MOE_T);

    // zero the accumulated output and the expert counters (graph-capturable)
    hipMemsetAsync(d_out, 0, (size_t)out_size * sizeof(float), stream);
    hipMemsetAsync(cnt, 0, MOE_E * sizeof(int), stream);

    moe_router_kernel<<<MOE_T / 8, 256, 0, stream>>>(
        x, gate_w, logits, cnt, tok_idx, tok_w);

    dim3 grid(MOE_H / NT, MOE_T / MT, MOE_E);
    moe_expert_kernel<<<grid, THREADS, 0, stream>>>(
        x, expert_w, cnt, tok_idx, tok_w, out);
}
